// StandardSelfAttention_7009386627368
// MI455X (gfx1250) — matrix operand order repaired and checked by reading, not compiled, not run on hardware
//
#include <hip/hip_runtime.h>
#include <math.h>

typedef __attribute__((ext_vector_type(16))) _Float16 v16h;
typedef __attribute__((ext_vector_type(8)))  _Float16 v8h;
typedef __attribute__((ext_vector_type(8)))  float    v8f;
typedef __attribute__((ext_vector_type(4)))  float    v4f;

constexpr int BB = 2, HH = 16, TT = 2048, DD = 64;
constexpr int BM = 128;   // query rows per block = 8 waves x 16
constexpr int ST = 32;    // key tile (one WMMA K for gemm2)
constexpr float TAU_RC = 0.02f, TAU_REF = 0.002f;
constexpr float INV_TEMP = 0.125f; // 1/sqrt(64)

// Async DMA of one 32x64 f32 V tile (8 KB) global -> LDS via the CDNA5
// ASYNCcnt path. 256 threads x 2 x b128 (4 floats). Inline asm because the
// async-to-LDS builtin arity differs across toolchains.
__device__ __forceinline__ void async_copy_vtile(const float* gsrc, float* ldsDst, int tid) {
  #pragma unroll
  for (int c = 0; c < 2; ++c) {
    const int elt = (c * 256 + tid) * 4;                    // float index in tile
    const uint64_t gaddr = (uint64_t)(uintptr_t)(gsrc + elt);
    // generic pointers to __shared__ carry the LDS byte offset in addr[31:0]
    const uint32_t laddr = (uint32_t)(uintptr_t)(ldsDst + elt);
    asm volatile("global_load_async_to_lds_b128 %0, %1, off"
                 :: "v"(laddr), "v"(gaddr) : "memory");
  }
}

__global__ __launch_bounds__(256)
void lif_attn_fwd(const float* __restrict__ Q, const float* __restrict__ V,
                  const float* __restrict__ E, const float* __restrict__ gain,
                  const float* __restrict__ bias, float* __restrict__ Out)
{
  // sP: per-wave 16x32 probability tile (C-layout -> A-layout transpose buffer)
  __shared__ _Float16 sP[8][16][ST];
  // sV: V tile transposed [d][k] so B-operand reads are contiguous ds_load_b128
  __shared__ _Float16 sV[DD][ST];
  // sVf32: double-buffered raw V tiles filled by the async DMA path
  __shared__ float sVf32[2][ST][DD];

  const int nQB  = TT / BM;
  const int qblk = blockIdx.x % nQB;
  const int bh   = blockIdx.x / nQB;
  const int h    = bh % HH;

  const size_t qkvOff = (size_t)bh * TT * DD;
  const float* Qb = Q + qkvOff;
  const float* Vb = V + qkvOff;
  const float* Eb = E + (size_t)h * TT * DD;
  const float* Gb = gain + (size_t)h * TT;
  const float* Bb = bias + (size_t)h * TT;
  float* Ob = Out + qkvOff;

  const int tid  = threadIdx.x;
  const int wave = tid >> 5;
  const int lane = tid & 31;
  const int ll   = lane & 15;   // low lane id (N for B/C, M for A)
  const int hi   = lane >> 4;   // half-wave select

  const int q0  = qblk * BM;
  const int qr0 = q0 + wave * 16;

  // ---- Q tile for this wave in A-operand layout: 2 K-chunks of 32 ----
  // A 16x32 f16: lane<16 row M=lane, halves 0-15 = K hi*16+0..15 (contiguous)
  v16h aq[2];
  {
    const float* qrow = Qb + (size_t)(qr0 + ll) * DD;
    #pragma unroll
    for (int kc = 0; kc < 2; ++kc) {
      const int c0 = kc*32 + hi*16;
      const int c1 = kc*32 + hi*16 + 8;
      v4f f0 = *(const v4f*)(qrow + c0);
      v4f f1 = *(const v4f*)(qrow + c0 + 4);
      v4f f2 = *(const v4f*)(qrow + c1);
      v4f f3 = *(const v4f*)(qrow + c1 + 4);
      #pragma unroll
      for (int j = 0; j < 4; ++j) {
        aq[kc][j]      = (_Float16)f0[j];
        aq[kc][4 + j]  = (_Float16)f1[j];
        aq[kc][8 + j]  = (_Float16)f2[j];
        aq[kc][12 + j] = (_Float16)f3[j];
      }
    }
  }

  v8f oacc[4] = {};            // O accumulators, 4 d-subtiles of 16
  float rowmax[8], rowsum[8];  // per C-reg row stats (row m = r + 8*hi)
  #pragma unroll
  for (int r = 0; r < 8; ++r) { rowmax[r] = -INFINITY; rowsum[r] = 0.0f; }

  const int nIter = (q0 + BM) / ST;       // causal: keys [0, q0+BM)

  // prologue: kick off the first V-tile DMA
  async_copy_vtile(Vb, &sVf32[0][0][0], tid);

  for (int it = 0; it < nIter; ++it) {
    const int s0  = it * ST;
    const int buf = it & 1;

    if (it + 1 < nIter)                   // speculative prefetch of next E tile
      __builtin_prefetch(Eb + (size_t)(s0 + ST + lane) * DD + wave * 8, 0, 0);

    // ---- gemm1: S = Q . E^T  (two 16-key sub-tiles) ----
    // B 32x16 f16: lane<16 -> K 0..15, lane>=16 -> K 16..31, halves contiguous in K
    v8f sacc[2] = {};
    #pragma unroll
    for (int nt = 0; nt < 2; ++nt) {
      #pragma unroll
      for (int kc = 0; kc < 2; ++kc) {
        const float* erow = Eb + (size_t)(s0 + nt*16 + ll) * DD + kc*32 + hi*16;
        v4f e0 = *(const v4f*)(erow);
        v4f e1 = *(const v4f*)(erow + 4);
        v4f e2 = *(const v4f*)(erow + 8);
        v4f e3 = *(const v4f*)(erow + 12);
        v16h be;
        #pragma unroll
        for (int j = 0; j < 4; ++j) {
          be[j]      = (_Float16)e0[j];
          be[4 + j]  = (_Float16)e1[j];
          be[8 + j]  = (_Float16)e2[j];
          be[12 + j] = (_Float16)e3[j];
        }
        sacc[nt] = __builtin_amdgcn_wmma_f32_16x16x32_f16(
            false, aq[kc], false, be, (short)0, sacc[nt], false, false);
      }
    }

    // ---- LIF rate + causal mask (C layout: n = ll is the key index) ----
    float st0[8], st1[8];
    {
      const int   sg0 = s0 + ll, sg1 = s0 + 16 + ll;
      const float g0 = Gb[sg0], b0 = Bb[sg0];
      const float g1 = Gb[sg1], b1 = Bb[sg1];
      #pragma unroll
      for (int r = 0; r < 8; ++r) {
        const int qr = qr0 + r + 8*hi;
        const float I0 = g0 * sacc[0][r] + b0;
        const float I1 = g1 * sacc[1][r] + b1;
        const float r0 = (I0 > 1.0f + 1e-7f)
            ? INV_TEMP / (TAU_REF - TAU_RC * log1pf(-1.0f / I0)) : 0.0f;
        const float r1 = (I1 > 1.0f + 1e-7f)
            ? INV_TEMP / (TAU_REF - TAU_RC * log1pf(-1.0f / I1)) : 0.0f;
        st0[r] = (qr >= sg0) ? r0 : -INFINITY;
        st1[r] = (qr >= sg1) ? r1 : -INFINITY;
      }
    }

    // ---- online softmax: rows live across 16 lanes of a half-wave ----
    #pragma unroll
    for (int r = 0; r < 8; ++r) {
      float mloc = fmaxf(st0[r], st1[r]);
      #pragma unroll
      for (int off = 1; off < 16; off <<= 1)
        mloc = fmaxf(mloc, __shfl_xor(mloc, off, 32));
      const float mnew = fmaxf(rowmax[r], mloc);
      const float corr = __expf(rowmax[r] - mnew);
      rowmax[r] = mnew;
      const float p0 = __expf(st0[r] - mnew);
      const float p1 = __expf(st1[r] - mnew);
      float ps = p0 + p1;
      #pragma unroll
      for (int off = 1; off < 16; off <<= 1)
        ps += __shfl_xor(ps, off, 32);
      rowsum[r] = rowsum[r] * corr + ps;
      #pragma unroll
      for (int n4 = 0; n4 < 4; ++n4) oacc[n4][r] *= corr;
      // scatter P to wave-private LDS in [m][k] layout (transpose staging)
      sP[wave][r + 8*hi][ll]      = (_Float16)p0;
      sP[wave][r + 8*hi][16 + ll] = (_Float16)p1;
    }

    // ---- retire this tile's async DMA, make LDS writes visible block-wide ----
    asm volatile("s_wait_asynccnt 0x0" ::: "memory");
    __syncthreads();

    // kick off next tile's DMA into the other buffer (overlaps transpose,
    // gemm2 and the next iteration's gemm1/softmax)
    if (it + 1 < nIter)
      async_copy_vtile(Vb + (size_t)(s0 + ST) * DD, &sVf32[buf ^ 1][0][0], tid);

    // cooperative transpose + f32->f16 convert into WMMA-friendly sV[d][k]
    #pragma unroll
    for (int i2 = tid; i2 < ST * DD; i2 += 256) {
      const int k = i2 >> 6, d = i2 & 63;
      sV[d][k] = (_Float16)sVf32[buf][k][d];
    }
    __syncthreads();           // sV ready for all waves

    // ---- gemm2: O += P . Vtile ----
    v16h ap;                   // A-layout read of P (wave-private, DS in-order)
    {
      const _Float16* prow = &sP[wave][ll][0];
      v8h plo = *(const v8h*)(prow + hi*8);
      v8h phi = *(const v8h*)(prow + 16 + hi*8);
      #pragma unroll
      for (int j = 0; j < 8; ++j) { ap[j] = plo[j]; ap[8 + j] = phi[j]; }
    }
    #pragma unroll
    for (int n4 = 0; n4 < 4; ++n4) {
      const _Float16* vrow = &sV[n4*16 + ll][0] + hi*8;
      v8h b0 = *(const v8h*)(vrow);
      v8h b1 = *(const v8h*)(vrow + 16);
      v16h bv;
      #pragma unroll
      for (int j = 0; j < 8; ++j) { bv[j] = b0[j]; bv[8 + j] = b1[j]; }
      oacc[n4] = __builtin_amdgcn_wmma_f32_16x16x32_f16(
          false, ap, false, bv, (short)0, oacc[n4], false, false);
    }
  }

  // ---- epilogue: O / rowsum ----
  #pragma unroll
  for (int r = 0; r < 8; ++r) {
    const int qr = qr0 + r + 8*hi;
    const float inv = 1.0f / rowsum[r];
    #pragma unroll
    for (int n4 = 0; n4 < 4; ++n4)
      Ob[(size_t)qr * DD + n4*16 + ll] = oacc[n4][r] * inv;
  }
}

extern "C" void kernel_launch(void* const* d_in, const int* in_sizes, int n_in,
                              void* d_out, int out_size, void* d_ws, size_t ws_size,
                              hipStream_t stream) {
  const float* Q    = (const float*)d_in[0];
  // d_in[1] is K: unused by the reference computation
  const float* V    = (const float*)d_in[2];
  const float* Ehat = (const float*)d_in[3];
  const float* g    = (const float*)d_in[4];
  const float* b    = (const float*)d_in[5];
  float* out = (float*)d_out;

  dim3 grid(BB * HH * (TT / BM));   // 2*16*16 = 512 blocks
  dim3 block(256);                  // 8 wave32s
  lif_attn_fwd<<<grid, block, 0, stream>>>(Q, V, Ehat, g, b, out);
}
